// DefaultAttentionModule_82240033784368
// MI455X (gfx1250) — matrix operand order repaired and checked by reading, not compiled, not run on hardware
//
#include <hip/hip_runtime.h>

// ---------------------------------------------------------------------------
// MLP (Linear -> BatchNorm(batch stats) -> ReLU -> Linear) + per-bag softmax
// for MI455X / gfx1250.
//
// GEMM1: split-fp16 WMMA (V_WMMA_F32_16X16X32_F16, Markidis x = hi + lo,
// 3 products, fp32 accumulate -> fp32-fidelity at f16 matrix rate).
// Roofline: 3*137 = 411 f16-GFLOP at ~5 PFLOP/s dense f16 ~= 82us vs ~70us
// HBM floor (1.07 GB X + 0.54 GB h traffic at 23.3 TB/s). Each wave computes
// a 16x32 tile (2 accumulators) so one A-fragment feeds 6 WMMAs, keeping the
// LDS pipe (2 ds_load_b128 per WMMA) from capping below the HBM floor.
//
// The per-bag softmax pass stages h chunks into LDS with the Tensor Data
// Mover (tensor_load_to_lds + s_wait_tensorcnt), 6-arg clang-23 builtin form.
// ---------------------------------------------------------------------------

typedef _Float16 v16h __attribute__((ext_vector_type(16)));
typedef _Float16 v8h  __attribute__((ext_vector_type(8)));
typedef _Float16 v4h  __attribute__((ext_vector_type(4)));
typedef _Float16 v2h  __attribute__((ext_vector_type(2)));
typedef float    v8f  __attribute__((ext_vector_type(8)));
typedef unsigned int u32x4 __attribute__((ext_vector_type(4)));
typedef int          i32x8 __attribute__((ext_vector_type(8)));
typedef int          i32x4 __attribute__((ext_vector_type(4)));

#define H    256      // hidden dim
#define DIN  1024     // input dim
#define KB   64       // K staged through LDS per block
#define BM   128      // rows per workgroup (one 16-row tile per wave)
#define BN   32       // cols per workgroup (two 16-col tiles per wave)
#define PA   72       // LDS stride (halves) for A tiles: 16B-aligned chunks
#define PB   72       // LDS stride (halves) for transposed B tiles
#define RCH  64       // rows per TDM chunk in the bag kernel

// ---------------------------------------------------------------------------
__global__ __launch_bounds__(256) void zero_stats_kernel(float* stats) {
    int t = blockIdx.x * blockDim.x + threadIdx.x;
    if (t < 2 * H) stats[t] = 0.0f;
}

// ---------------------------------------------------------------------------
// h = X @ W1 + b1 (split-f16 WMMA, fp32 accumulate), plus per-column stats.
__global__ __launch_bounds__(256) void gemm1_kernel(
    const float* __restrict__ X,    // [N, DIN]
    const float* __restrict__ W1,   // [DIN, H]
    const float* __restrict__ b1,   // [H]
    float* __restrict__ h,          // [N, H]
    float* __restrict__ colsum,     // [H]
    float* __restrict__ colsq)      // [H]
{
    __shared__ _Float16 lAhi[BM * PA];   // A tile, high f16 part [row][k]
    __shared__ _Float16 lAlo[BM * PA];   // A tile, low  f16 part
    __shared__ _Float16 lBhi[BN * PB];   // B tile TRANSPOSED [n][k], high part
    __shared__ _Float16 lBlo[BN * PB];   // B tile TRANSPOSED [n][k], low part
    __shared__ float sSum[BN];
    __shared__ float sSq[BN];

    const int tid  = threadIdx.x;
    const int lane = tid & 31;
    const int wave = tid >> 5;               // 8 waves = 8 row tiles
    const int rowBase = blockIdx.y * BM;
    const int colBase = blockIdx.x * BN;

    if (tid < BN) { sSum[tid] = 0.0f; sSq[tid] = 0.0f; }

    v8f c0 = {};                             // 16x16 fp32 accumulators:
    v8f c1 = {};                             // cols [0..15] and [16..31]

    // A fragment (16-bit A 16x32): lanes 0-15: M=lane, K chunks {0..7,16..23};
    // lanes 16-31: same M, K chunks shifted by +8.
    const int m    = lane & 15;
    const int nn   = lane & 15;
    const int aoff = (lane >> 4) << 3;       // 0 or 8 halves
    // B fragment (32x16): same K chunks as A: lanes 0-15 K {0..7,16..23}; lanes 16-31 +8.
    const int boff = (lane >> 4) << 3;       // 0 or 8 halves

    const _Float16* pAhi = &lAhi[(wave * 16 + m) * PA + aoff];
    const _Float16* pAlo = &lAlo[(wave * 16 + m) * PA + aoff];
    const _Float16* pB0h = &lBhi[nn * PB + boff];              // N = 0..15
    const _Float16* pB0l = &lBlo[nn * PB + boff];
    const _Float16* pB1h = &lBhi[(16 + nn) * PB + boff];       // N = 16..31
    const _Float16* pB1l = &lBlo[(16 + nn) * PB + boff];

    for (int kb = 0; kb < DIN; kb += KB) {
        __syncthreads();
        // ---- stage A tile 128x64 fp32 -> f16 hi/lo (8 float4 per thread) ----
        #pragma unroll
        for (int i = 0; i < 8; ++i) {
            int id = tid + 256 * i;
            int r  = id >> 4;                // 0..127
            int kq = (id & 15) << 2;         // 0..60
            float4 v = *(const float4*)(X + (size_t)(rowBase + r) * DIN + kb + kq);
            v4h hi4, lo4;
            const float* vf = (const float*)&v;
            #pragma unroll
            for (int e = 0; e < 4; ++e) {
                _Float16 hi = (_Float16)vf[e];
                hi4[e] = hi;
                lo4[e] = (_Float16)(vf[e] - (float)hi);
            }
            *(v4h*)&lAhi[r * PA + kq] = hi4;
            *(v4h*)&lAlo[r * PA + kq] = lo4;
        }
        // ---- stage B tile 64x32 fp32 -> f16 hi/lo, transposed to [n][k]:
        //      two consecutive k rows per thread -> packed half2 stores ----
        {
            int kk = (tid >> 3) << 1;        // 0,2,..,62
            int nq = (tid & 7) << 2;         // 0..28
            const float* p0 = W1 + (size_t)(kb + kk) * H + colBase + nq;
            float4 v0 = *(const float4*)(p0);
            float4 v1 = *(const float4*)(p0 + H);
            const float* f0 = (const float*)&v0;
            const float* f1 = (const float*)&v1;
            #pragma unroll
            for (int e = 0; e < 4; ++e) {
                _Float16 h0 = (_Float16)f0[e];
                _Float16 h1 = (_Float16)f1[e];
                v2h hi2 = {h0, h1};
                v2h lo2 = {(_Float16)(f0[e] - (float)h0),
                           (_Float16)(f1[e] - (float)h1)};
                *(v2h*)&lBhi[(nq + e) * PB + kk] = hi2;
                *(v2h*)&lBlo[(nq + e) * PB + kk] = lo2;
            }
        }
        __syncthreads();

        // prefetch next A K-tile (global_prefetch_b8)
        if (kb + KB < DIN) {
            __builtin_prefetch(X + (size_t)(rowBase + (tid >> 1)) * DIN
                                 + kb + KB + ((tid & 1) << 5), 0, 0);
        }

        #pragma unroll
        for (int ks = 0; ks < KB; ks += 32) {
            union { v16h v; v8h h8[2]; } ahi, alo, b0h, b0l, b1h, b1l;
            ahi.h8[0] = *(const v8h*)(pAhi + ks);
            ahi.h8[1] = *(const v8h*)(pAhi + ks + 16);
            alo.h8[0] = *(const v8h*)(pAlo + ks);
            alo.h8[1] = *(const v8h*)(pAlo + ks + 16);
            b0h.h8[0] = *(const v8h*)(pB0h + ks);
            b0h.h8[1] = *(const v8h*)(pB0h + ks + 16);
            b0l.h8[0] = *(const v8h*)(pB0l + ks);
            b0l.h8[1] = *(const v8h*)(pB0l + ks + 16);
            b1h.h8[0] = *(const v8h*)(pB1h + ks);
            b1h.h8[1] = *(const v8h*)(pB1h + ks + 16);
            b1l.h8[0] = *(const v8h*)(pB1l + ks);
            b1l.h8[1] = *(const v8h*)(pB1l + ks + 16);
            // D = Ahi*Bhi + Ahi*Blo + Alo*Bhi + C  per col tile (err ~2^-22)
            c0 = __builtin_amdgcn_wmma_f32_16x16x32_f16(
                     false, ahi.v, false, b0h.v, (short)0, c0, false, false);
            c0 = __builtin_amdgcn_wmma_f32_16x16x32_f16(
                     false, ahi.v, false, b0l.v, (short)0, c0, false, false);
            c0 = __builtin_amdgcn_wmma_f32_16x16x32_f16(
                     false, alo.v, false, b0h.v, (short)0, c0, false, false);
            c1 = __builtin_amdgcn_wmma_f32_16x16x32_f16(
                     false, ahi.v, false, b1h.v, (short)0, c1, false, false);
            c1 = __builtin_amdgcn_wmma_f32_16x16x32_f16(
                     false, ahi.v, false, b1l.v, (short)0, c1, false, false);
            c1 = __builtin_amdgcn_wmma_f32_16x16x32_f16(
                     false, alo.v, false, b1h.v, (short)0, c1, false, false);
        }
    }

    // ---- epilogue: + b1, store h, per-column stats ----
    const int gcol0 = colBase + nn;
    const int gcol1 = colBase + 16 + nn;
    const float bias0 = b1[gcol0];
    const float bias1 = b1[gcol1];
    float ps0 = 0.0f, pq0 = 0.0f, ps1 = 0.0f, pq1 = 0.0f;
    #pragma unroll
    for (int v = 0; v < 8; ++v) {
        int rloc = (lane < 16) ? v : (v + 8);    // C layout: lanes 16-31 = M+8
        size_t grow = (size_t)(rowBase + wave * 16 + rloc);
        float val0 = c0[v] + bias0;
        float val1 = c1[v] + bias1;
        h[grow * H + gcol0] = val0;
        h[grow * H + gcol1] = val1;
        ps0 += val0; pq0 += val0 * val0;
        ps1 += val1; pq1 += val1 * val1;
    }
    atomicAdd(&sSum[nn],      ps0);
    atomicAdd(&sSq [nn],      pq0);
    atomicAdd(&sSum[16 + nn], ps1);
    atomicAdd(&sSq [16 + nn], pq1);
    __syncthreads();
    if (tid < BN)            atomicAdd(&colsum[colBase + tid],      sSum[tid]);
    else if (tid < 2 * BN)   atomicAdd(&colsq [colBase + tid - BN], sSq[tid - BN]);
}

// ---------------------------------------------------------------------------
__global__ __launch_bounds__(256) void finalize_kernel(
    const float* __restrict__ colsum, const float* __restrict__ colsq,
    const float* __restrict__ gamma,  const float* __restrict__ beta,
    float* __restrict__ scale, float* __restrict__ shift, float Nf)
{
    int t = threadIdx.x;
    if (t < H) {
        float mean = colsum[t] / Nf;
        float var  = colsq[t] / Nf - mean * mean;
        float sc   = gamma[t] * rsqrtf(var + 1e-5f);
        scale[t] = sc;
        shift[t] = beta[t] - mean * sc;
    }
}

// ---------------------------------------------------------------------------
// Tensor Data Mover: 2D tile Global -> LDS. D# packed per ISA 8.3/8.4:
// group0 = {count=1 | lds_addr | global_addr | type=2}, group1 carries
// data_size / pad / dims / strides. clang-23 lane: 6-arg builtin.
__device__ __forceinline__ void tdm_load_2d(
    unsigned int lds_byte_off, const void* gptr,
    unsigned int dim0, unsigned int dim1, unsigned int stride0,
    unsigned int tile0, unsigned int tile1)
{
    unsigned long long ga = (unsigned long long)(uintptr_t)gptr;
    u32x4 g0;
    g0[0] = 1u;                                           // count=1, user D#
    g0[1] = lds_byte_off;                                 // lds_addr
    g0[2] = (unsigned int)ga;                             // global_addr lo
    g0[3] = (unsigned int)((ga >> 32) & 0x01FFFFFFu)      // global_addr hi
          | (2u << 30);                                   // type = 2 (image)
    i32x8 g1;
    g1[0] = (int)(2u << 16);                              // data_size=4B, no pad
    g1[1] = (int)((dim0 & 0xFFFFu) << 16);                // tensor_dim0[15:0]
    g1[2] = (int)((dim0 >> 16) | ((dim1 & 0xFFFFu) << 16));
    g1[3] = (int)((dim1 >> 16) | ((tile0 & 0xFFFFu) << 16));
    g1[4] = (int)(tile1 & 0xFFFFu);                       // tile_dim1, tile_dim2=0
    g1[5] = (int)stride0;                                 // tensor_dim0_stride lo
    g1[6] = 0;
    g1[7] = 0;
    i32x4 z4 = {0, 0, 0, 0};
    i32x8 z8 = {0, 0, 0, 0, 0, 0, 0, 0};
    __builtin_amdgcn_tensor_load_to_lds(g0, g1, z4, z4, z8, 0);
}

// One workgroup per bag: scores = relu(h*scale+shift) @ W2 + b2, then
// in-LDS stable softmax. h rows are TDM-staged into LDS in 64-row chunks.
__global__ __launch_bounds__(256) void bag_softmax_kernel(
    const float* __restrict__ h, const long long* __restrict__ bag_sizes,
    const float* __restrict__ scale, const float* __restrict__ shift,
    const float* __restrict__ W2, const float* __restrict__ b2,
    float* __restrict__ out)
{
    __shared__ float lH[RCH * H];        // 64 KB TDM landing buffer
    __shared__ float sSc[H], sSh[H], sW2[H];
    __shared__ float sScores[2048];
    __shared__ float red[8];
    __shared__ int   sStart, sCount;
    __shared__ float sM, sZ;

    const int tid  = threadIdx.x;
    const int lane = tid & 31;
    const int wave = tid >> 5;
    const int bag  = blockIdx.x;

    if (tid < H) { sSc[tid] = scale[tid]; sSh[tid] = shift[tid]; sW2[tid] = W2[tid]; }
    if (tid == 0) {
        long long s = 0;
        for (int i = 0; i < bag; ++i) s += bag_sizes[i];
        sStart = (int)s;
        long long cnt = bag_sizes[bag];
        sCount = (int)(cnt > 2048 ? 2048 : cnt);
    }
    __syncthreads();
    const int start = sStart, count = sCount;
    const float bb = b2[0];

    for (int base = 0; base < count; base += RCH) {
        const int rows = (count - base) < RCH ? (count - base) : RCH;
        __syncthreads();           // previous chunk fully consumed
        if (wave == 0) {           // one wave issues the DMA (EXEC ignored)
            tdm_load_2d((unsigned int)(uintptr_t)&lH[0],
                        h + (size_t)(start + base) * H,
                        /*dim0=*/H, /*dim1=*/(unsigned)rows, /*stride0=*/H,
                        /*tile0=*/H, /*tile1=*/(unsigned)rows);
            __builtin_amdgcn_s_wait_tensorcnt(0);
        }
        __syncthreads();           // LDS chunk visible to all waves

        // one row per wave; lane covers 8 contiguous columns
        for (int i = wave; i < rows; i += 8) {
            const float* hr = lH + i * H;
            float acc = 0.0f;
            #pragma unroll
            for (int q = 0; q < 2; ++q) {
                float4 hv = *(const float4*)(hr + lane * 8 + q * 4);
                #pragma unroll
                for (int e = 0; e < 4; ++e) {
                    int col = lane * 8 + q * 4 + e;
                    float x = ((const float*)&hv)[e] * sSc[col] + sSh[col];
                    x = x > 0.0f ? x : 0.0f;
                    acc += x * sW2[col];
                }
            }
            #pragma unroll
            for (int o = 16; o > 0; o >>= 1) acc += __shfl_down(acc, o, 32);
            if (lane == 0) sScores[base + i] = acc + bb;
        }
    }
    __syncthreads();

    // block max
    float m = -3.402823466e38f;
    for (int i = tid; i < count; i += 256) m = fmaxf(m, sScores[i]);
    #pragma unroll
    for (int o = 16; o > 0; o >>= 1) m = fmaxf(m, __shfl_down(m, o, 32));
    if (lane == 0) red[wave] = m;
    __syncthreads();
    if (tid == 0) {
        float mm = red[0];
        for (int i = 1; i < 8; ++i) mm = fmaxf(mm, red[i]);
        sM = mm;
    }
    __syncthreads();
    const float mMax = sM;

    // exp + block sum
    float z = 0.0f;
    for (int i = tid; i < count; i += 256) {
        float e = __expf(sScores[i] - mMax);
        sScores[i] = e;
        z += e;
    }
    #pragma unroll
    for (int o = 16; o > 0; o >>= 1) z += __shfl_down(z, o, 32);
    if (lane == 0) red[wave] = z;
    __syncthreads();
    if (tid == 0) {
        float zz = 0.0f;
        for (int i = 0; i < 8; ++i) zz += red[i];
        sZ = zz;
    }
    __syncthreads();
    const float inv = 1.0f / sZ;
    for (int i = tid; i < count; i += 256) out[start + i] = sScores[i] * inv;
}

// ---------------------------------------------------------------------------
extern "C" void kernel_launch(void* const* d_in, const int* in_sizes, int n_in,
                              void* d_out, int out_size, void* d_ws, size_t ws_size,
                              hipStream_t stream) {
    const float*     features  = (const float*)d_in[0];
    const long long* bag_sizes = (const long long*)d_in[1];
    const float*     W1        = (const float*)d_in[2];
    const float*     b1        = (const float*)d_in[3];
    const float*     gamma     = (const float*)d_in[4];
    const float*     beta      = (const float*)d_in[5];
    const float*     W2        = (const float*)d_in[6];
    const float*     b2        = (const float*)d_in[7];
    float*           out       = (float*)d_out;

    const int N      = in_sizes[0] / DIN;   // 262144
    const int n_bags = in_sizes[1];         // 128

    // workspace: h[N*H] | colsum[H] | colsq[H] | scale[H] | shift[H]
    float* h      = (float*)d_ws;
    float* stats  = h + (size_t)N * H;
    float* colsum = stats;
    float* colsq  = stats + H;
    float* scale  = stats + 2 * H;
    float* shift  = stats + 3 * H;

    zero_stats_kernel<<<2, 256, 0, stream>>>(stats);

    dim3 grid(H / BN, N / BM);
    gemm1_kernel<<<grid, 256, 0, stream>>>(features, W1, b1, h, colsum, colsq);

    finalize_kernel<<<1, 256, 0, stream>>>(colsum, colsq, gamma, beta, scale, shift, (float)N);

    bag_softmax_kernel<<<n_bags, 256, 0, stream>>>(h, bag_sizes, scale, shift, W2, b2, out);
}
